// MultiHeadAttention_88441966559971
// MI455X (gfx1250) — compile-verified, matrix operand order checked by reading, not run on hardware
//
#include <hip/hip_runtime.h>
#include <hip/hip_bf16.h>
#include <math.h>

// MI455X / gfx1250, wave32. D = A*B + C via v_wmma_f32_16x16x32_bf16.
// Key fact: the reference's (B,S,D)->(B,H,S,dh) "reshape" is a raw view, so
// the head-major flat layout is IDENTICAL to the (g=b*S+s, d) flat layout:
//   ((g>>8)*SEQ + (g&255)*8 + (d>>4))*16 + (d&15) == g*128 + d.
// Q/K/O therefore live at their natural addresses; only V needs a transpose
// (dim-major) for the P@V WMMA B-operand.
#define EMB   128
#define HEADS 8
#define DH    16
#define SEQ   2048
#define KB    64                // keys per attention block
#define NBLK  (SEQ / KB)        // 32

typedef __attribute__((ext_vector_type(16))) __bf16 v16bf;
typedef __attribute__((ext_vector_type(8)))  __bf16 v8bf;
typedef __attribute__((ext_vector_type(8)))  float  v8f;
typedef __attribute__((ext_vector_type(4)))  int    v4i;

#if __has_builtin(__builtin_amdgcn_global_load_async_to_lds_b128) && \
    __has_builtin(__builtin_amdgcn_s_wait_asynccnt)
#define ASYNC_OK 1
#else
#define ASYNC_OK 0
#endif

static __device__ __forceinline__ __bf16 f2bf(float f) { return (__bf16)f; }

static __device__ __forceinline__ v16bf cat16(v8bf lo, v8bf hi) {
  v16bf r;
#pragma unroll
  for (int i = 0; i < 8; ++i) { r[i] = lo[i]; r[i + 8] = hi[i]; }
  return r;
}

static __device__ __forceinline__ v8bf zero8bf() {
  v8bf z;
#pragma unroll
  for (int i = 0; i < 8; ++i) z[i] = __builtin_bit_cast(__bf16, (unsigned short)0);
  return z;
}

static __device__ __forceinline__ v8f wmma_bf16(v16bf a, v16bf b, v8f c) {
  // (neg_a, A, neg_b, B, c_mod, C, reuse_a, reuse_b)
  return __builtin_amdgcn_wmma_f32_16x16x32_bf16(false, a, false, b, (short)0, c, false, false);
}

static __device__ __forceinline__ void copy16(void* lds, const void* g) {
#if ASYNC_OK
  // (global src, lds dst, imm offset, imm cpol) -- params are generic v4i*
  __builtin_amdgcn_global_load_async_to_lds_b128((v4i*)g, (v4i*)lds, 0, 0);
#else
  *(uint4*)lds = *(const uint4*)g;
#endif
}

// ---------------------------------------------------------------------------
// Kernel 0: convert the four 128x128 weight matrices fp32 -> bf16 in workspace
// ---------------------------------------------------------------------------
__global__ void k_convert_w(const float* __restrict__ Wq, const float* __restrict__ Wk,
                            const float* __restrict__ Wv, const float* __restrict__ Wo,
                            __bf16* __restrict__ out) {
  const int i = blockIdx.x * blockDim.x + threadIdx.x;
  if (i >= EMB * EMB) return;
  out[i]             = f2bf(Wq[i]);
  out[16384 + i]     = f2bf(Wk[i]);
  out[2 * 16384 + i] = f2bf(Wv[i]);
  out[3 * 16384 + i] = f2bf(Wo[i]);
}

// ---------------------------------------------------------------------------
// Kernel A: LayerNorm + QKV projections (xn @ W^T + b).
// Q,K stored at natural flat addresses (== head-major view, see header note);
// V stored transposed per head: vT[bh][d2][s2].
// ---------------------------------------------------------------------------
__global__ __launch_bounds__(256) void k_ln_qkv(
    const float* __restrict__ x, const float* __restrict__ ln_w, const float* __restrict__ ln_b,
    const __bf16* __restrict__ Wb,
    const float* __restrict__ bq, const float* __restrict__ bk, const float* __restrict__ bv,
    __bf16* __restrict__ qh, __bf16* __restrict__ kh, __bf16* __restrict__ vT) {
  __shared__ __attribute__((aligned(16))) __bf16 xn[128 * 136];  // 136 stride: bank deskew
  const int tid   = threadIdx.x;
  const int gbase = blockIdx.x * 128;  // base row in flattened (B*S)

  // --- LayerNorm: two threads per row -------------------------------------
  {
    const int r = tid >> 1, half = tid & 1;
    const float* xr = x + (size_t)(gbase + r) * EMB + half * 64;
    float vals[64];
    float s = 0.f;
#pragma unroll
    for (int i = 0; i < 16; ++i) {
      const float4 f = ((const float4*)xr)[i];
      vals[4 * i] = f.x; vals[4 * i + 1] = f.y; vals[4 * i + 2] = f.z; vals[4 * i + 3] = f.w;
      s += f.x + f.y + f.z + f.w;
    }
    s += __shfl_xor(s, 1, 32);
    const float mu = s * (1.f / 128.f);
    float vs = 0.f;
#pragma unroll
    for (int i = 0; i < 64; ++i) { const float d = vals[i] - mu; vs += d * d; }
    vs += __shfl_xor(vs, 1, 32);
    const float rstd = rsqrtf(vs * (1.f / 128.f) + 1e-5f);
#pragma unroll
    for (int i = 0; i < 64; ++i) {
      const int d = half * 64 + i;
      xn[r * 136 + d] = f2bf((vals[i] - mu) * rstd * ln_w[d] + ln_b[d]);
    }
  }
  __syncthreads();

  // --- QKV: 3 mats x 8 Mtiles x 8 Ntiles = 192 16x16 tiles over 8 waves ---
  const int wave = tid >> 5, lane = tid & 31;
  const int lhalf = lane & 15;
  const int sel   = (lane < 16) ? 0 : 8;   // K-chunk selector per A/B packing
  const int roff  = (lane >> 4) << 3;      // C-layout row offset for this half

  for (int t = wave; t < 192; t += 8) {
    const int mat = t >> 6, rem = t & 63, mt = rem >> 3, nt = rem & 7;
    const __bf16* W = Wb + mat * 16384;
    const int arow = mt * 16 + lhalf;
    const int d    = nt * 16 + lhalf;
    v8f acc = {};
#pragma unroll
    for (int kc = 0; kc < 4; ++kc) {
      const int k0 = kc * 32 + sel;
      v16bf A = cat16(*(const v8bf*)&xn[arow * 136 + k0],
                      *(const v8bf*)&xn[arow * 136 + k0 + 16]);
      v16bf B = cat16(*(const v8bf*)&W[d * 128 + k0],
                      *(const v8bf*)&W[d * 128 + k0 + 16]);  // col d of W^T = row d of W
      acc = wmma_bf16(A, B, acc);
    }
    const float bias = (mat == 0) ? bq[d] : (mat == 1) ? bk[d] : bv[d];
    const int rl0 = mt * 16 + roff;  // local row of acc[0]

    if (mat < 2) {
      // natural flat layout; one base, r*128-element immediate strides
      __bf16* dst = (mat == 0 ? qh : kh) + (size_t)(gbase + rl0) * EMB + d;
#pragma unroll
      for (int r = 0; r < 8; ++r) dst[r * EMB] = f2bf(acc[r] + bias);
    } else {
      // vT[bh][d&15][s2]: bh = g>>8 (workgroup-constant), s2 = (g&255)*8 + d/16
      __bf16* dst = vT + ((size_t)(gbase >> 8) * DH + (d & 15)) * SEQ +
                    ((gbase & 255) + rl0) * 8 + (d >> 4);
#pragma unroll
      for (int r = 0; r < 8; ++r) dst[r * 8] = f2bf(acc[r] + bias);
    }
  }
}

// ---------------------------------------------------------------------------
// Kernel B: flash attention per (b,h). One wave owns 16 query rows. K/V blocks
// (64 keys) are staged once per workgroup into LDS, double-buffered via
// async global->LDS (ASYNCcnt). Softmax stats via 16-lane shuffles; probs
// staged through LDS to re-layout C->A for the P@V WMMA.
// Post-softmax scale 1/sqrt(128) (reference quirk).
// ---------------------------------------------------------------------------
__global__ __launch_bounds__(256) void k_attn(
    const __bf16* __restrict__ qh, const __bf16* __restrict__ kh,
    const __bf16* __restrict__ vT, __bf16* __restrict__ oh) {
  __shared__ __attribute__((aligned(16))) __bf16 sK[2][KB * DH];  // key-major, 2x2KB
  __shared__ __attribute__((aligned(16))) __bf16 sV[2][DH * KB];  // dim-major, 2x2KB
  __shared__ __attribute__((aligned(16))) __bf16 sP[8][16 * KB];  // 16KB (per-wave 2KB)
  const int tid  = threadIdx.x;
  const int wave = tid >> 5, lane = tid & 31;
  const int lhalf = lane & 15;
  const int sel   = (lane < 16) ? 0 : 8;
  const int roff  = (lane >> 4) << 3;
  const int bh = blockIdx.x >> 4, qt = blockIdx.x & 15;
  const __bf16* qp = qh + (size_t)bh * SEQ * DH;
  const __bf16* kp = kh + (size_t)bh * SEQ * DH;
  const __bf16* vp = vT + (size_t)bh * DH * SEQ;
  const int qrow0 = qt * 128 + wave * 16;

  // Staging: 256 threads x one 16B chunk = 4KB (K block 2KB + V block 2KB);
  // exactly one async b128 instruction per wave per block.
  const bool isK = tid < 128;
  const int kchunk = tid;             // K: 128 chunks of 8 elems
  const int vd  = (tid - 128) >> 3;   // V: dim row
  const int vcc = (tid - 128) & 7;    // V: chunk within row

  auto stage = [&](int buf, int j) {
    if (isK) copy16(&sK[buf][kchunk * 8], kp + (size_t)j * DH + kchunk * 8);
    else     copy16(&sV[buf][vd * KB + vcc * 8], vp + (size_t)vd * SEQ + j + vcc * 8);
  };

  // A operand: Q rows, dh=16 real K values, zero-pad K=16..31
  v16bf Aq = cat16(*(const v8bf*)&qp[(qrow0 + lhalf) * DH + sel], zero8bf());

  v8f accO = {};
  float m[8], l[8];
#pragma unroll
  for (int r = 0; r < 8; ++r) { m[r] = -1e30f; l[r] = 0.f; }

  __bf16* myP = &sP[wave][0];

  stage(0, 0);  // prologue: prefetch first block

  for (int jb = 0; jb < NBLK; ++jb) {
    const int cur = jb & 1;
#if ASYNC_OK
    __builtin_amdgcn_s_wait_asynccnt(0);  // this wave's staging chunk landed
#endif
    __syncthreads();                       // all waves staged; prev buffer free
    if (jb + 1 < NBLK) stage(cur ^ 1, (jb + 1) * KB);

    // --- scores: 4 x (16x16) via WMMA, K padded 16->32 ---------------------
    v8f z = {};
    v8f sc[4];
#pragma unroll
    for (int k = 0; k < 4; ++k) {
      v16bf Bt = cat16(*(const v8bf*)&sK[cur][(k * 16 + lhalf) * DH + sel], zero8bf());
      sc[k] = wmma_bf16(Aq, Bt, z);
    }

    // --- streaming softmax over this 64-key block --------------------------
#pragma unroll
    for (int r = 0; r < 8; ++r) {
      float a0 = sc[0][r], a1 = sc[1][r], a2 = sc[2][r], a3 = sc[3][r];
      float mx = fmaxf(fmaxf(a0, a1), fmaxf(a2, a3));
#pragma unroll
      for (int msk = 1; msk < 16; msk <<= 1) mx = fmaxf(mx, __shfl_xor(mx, msk, 32));
      const float mnew  = fmaxf(m[r], mx);
      const float scale = __expf(m[r] - mnew);
      const float p0 = __expf(a0 - mnew), p1 = __expf(a1 - mnew);
      const float p2 = __expf(a2 - mnew), p3 = __expf(a3 - mnew);
      float rs = (p0 + p1) + (p2 + p3);
#pragma unroll
      for (int msk = 1; msk < 16; msk <<= 1) rs += __shfl_xor(rs, msk, 32);
      l[r] = l[r] * scale + rs;
      m[r] = mnew;
      accO[r] *= scale;  // rescale running O before adding this block
      __bf16* pp = &myP[(r + roff) * KB + lhalf];
      pp[0] = f2bf(p0); pp[16] = f2bf(p1); pp[32] = f2bf(p2); pp[48] = f2bf(p3);
    }
    // DS stores (C-layout) must land before DS reads (A-layout): wave-local sync
    asm volatile("s_wait_dscnt 0x0" ::: "memory");
    __builtin_amdgcn_wave_barrier();

    // --- O += P @ V : two 16x16x32 WMMAs -----------------------------------
    {
      const __bf16* pr0 = &myP[lhalf * KB];
      v16bf A0 = cat16(*(const v8bf*)&pr0[sel],      *(const v8bf*)&pr0[16 + sel]);
      v16bf A1 = cat16(*(const v8bf*)&pr0[32 + sel], *(const v8bf*)&pr0[48 + sel]);
      const __bf16* vr0 = &sV[cur][lhalf * KB];
      v16bf Bv0 = cat16(*(const v8bf*)&vr0[sel],      *(const v8bf*)&vr0[16 + sel]);
      v16bf Bv1 = cat16(*(const v8bf*)&vr0[32 + sel], *(const v8bf*)&vr0[48 + sel]);
      accO = wmma_bf16(A0, Bv0, accO);
      accO = wmma_bf16(A1, Bv1, accO);
    }
    __builtin_amdgcn_wave_barrier();
  }

  const float dscale = 0.08838834764831845f;  // 1/sqrt(128), applied AFTER softmax
  __bf16* od = oh + ((size_t)bh * SEQ + qrow0 + roff) * DH + lhalf;
#pragma unroll
  for (int r = 0; r < 8; ++r) od[r * DH] = f2bf(accO[r] * (dscale / l[r]));
}

// ---------------------------------------------------------------------------
// Kernel C: output projection (att @ Wo^T + bo) + residual. Because the view
// permutation is a memory identity, att rows are just oh at g*128+d: the
// "gather" is a linear 32KB block copy into LDS.
// ---------------------------------------------------------------------------
__global__ __launch_bounds__(256) void k_out_proj(
    const __bf16* __restrict__ oh, const __bf16* __restrict__ Wb,
    const float* __restrict__ bo, const float* __restrict__ x, float* __restrict__ out) {
  __shared__ __attribute__((aligned(16))) __bf16 att[128 * 136];
  const int tid   = threadIdx.x;
  const int gbase = blockIdx.x * 128;
  {
    const int r = tid >> 1, half = tid & 1;
    const uint4* src = (const uint4*)(oh + (size_t)(gbase + r) * EMB + half * 64);
    uint4* dst = (uint4*)&att[r * 136 + half * 64];
#pragma unroll
    for (int i = 0; i < 8; ++i) dst[i] = src[i];
  }
  __syncthreads();

  const int wave = tid >> 5, lane = tid & 31;
  const int lhalf = lane & 15;
  const int sel   = (lane < 16) ? 0 : 8;
  const int roff  = (lane >> 4) << 3;
  const __bf16* Wo = Wb + 3 * 16384;

  for (int t = wave; t < 64; t += 8) {
    const int mt = t >> 3, nt = t & 7;
    const int arow = mt * 16 + lhalf;
    const int d    = nt * 16 + lhalf;
    v8f acc = {};
#pragma unroll
    for (int kc = 0; kc < 4; ++kc) {
      const int k0 = kc * 32 + sel;
      v16bf A = cat16(*(const v8bf*)&att[arow * 136 + k0],
                      *(const v8bf*)&att[arow * 136 + k0 + 16]);
      v16bf B = cat16(*(const v8bf*)&Wo[d * 128 + k0],
                      *(const v8bf*)&Wo[d * 128 + k0 + 16]);
      acc = wmma_bf16(A, B, acc);
    }
    const float bias = bo[d];
    const size_t base = (size_t)(gbase + mt * 16 + roff) * EMB + d;
#pragma unroll
    for (int r = 0; r < 8; ++r)
      out[base + r * EMB] = acc[r] + bias + x[base + r * EMB];
  }
}

// ---------------------------------------------------------------------------
extern "C" void kernel_launch(void* const* d_in, const int* in_sizes, int n_in,
                              void* d_out, int out_size, void* d_ws, size_t ws_size,
                              hipStream_t stream) {
  const float* x    = (const float*)d_in[0];
  const float* ln_w = (const float*)d_in[1];
  const float* ln_b = (const float*)d_in[2];
  const float* Wq   = (const float*)d_in[3];
  const float* bq   = (const float*)d_in[4];
  const float* Wk   = (const float*)d_in[5];
  const float* bk   = (const float*)d_in[6];
  const float* Wv   = (const float*)d_in[7];
  const float* bv   = (const float*)d_in[8];
  const float* Wo   = (const float*)d_in[9];
  const float* bo   = (const float*)d_in[10];

  // Workspace layout (bytes):
  //   [0, 128K)          : bf16 weights Wq|Wk|Wv|Wo
  //   [128K, +4M)        : qh  bf16 (flat == [bh][s2][d2])
  //   [128K+4M, +4M)     : kh  bf16 (flat == [bh][s2][d2])
  //   [128K+8M, +4M)     : vT  bf16 [bh][d2][s2] (transposed)
  //   [128K+12M, +4M)    : oh  bf16 (flat == [bh][s2][d2])
  char* ws = (char*)d_ws;
  __bf16* Wb = (__bf16*)ws;
  __bf16* qh = (__bf16*)(ws + (128u << 10));
  __bf16* kh = (__bf16*)(ws + (128u << 10) + (4u << 20));
  __bf16* vT = (__bf16*)(ws + (128u << 10) + (8u << 20));
  __bf16* oh = (__bf16*)(ws + (128u << 10) + (12u << 20));

  k_convert_w<<<64, 256, 0, stream>>>(Wq, Wk, Wv, Wo, Wb);
  k_ln_qkv<<<128, 256, 0, stream>>>(x, ln_w, ln_b, Wb, bq, bk, bv, qh, kh, vT);
  k_attn<<<1024, 256, 0, stream>>>(qh, kh, vT, oh);
  k_out_proj<<<128, 256, 0, stream>>>(oh, Wb, bo, x, (float*)d_out);
}
